// GIN_75608604279030
// MI455X (gfx1250) — matrix operand order repaired and checked by reading, not compiled, not run on hardware
//
#include <hip/hip_runtime.h>

#define NN   100000
#define EE   1600000
#define HH   128
#define DIN_ 64

typedef __attribute__((ext_vector_type(16))) __bf16 v16bf;
typedef __attribute__((ext_vector_type(8)))  float  v8f;
typedef unsigned short ushort_t;

union BfVec {
  ushort_t s[16];
  uint4    q[2];
  v16bf    v;
};

__device__ __forceinline__ ushort_t f2bf(float f) {
  __bf16 b = (__bf16)f;
  return *reinterpret_cast<ushort_t*>(&b);
}

__device__ __forceinline__ unsigned int pack2bf(float a, float b) {
  return (unsigned int)f2bf(a) | ((unsigned int)f2bf(b) << 16);
}

// ---------------- edge aggregation: agg[dst] += x[src] ----------------
// one wave32 per edge; lane handles K/32 consecutive floats
template <int K>
__global__ void __launch_bounds__(256) scatter_agg(const float* __restrict__ x,
                                                   const int* __restrict__ src,
                                                   const int* __restrict__ dst,
                                                   float* __restrict__ agg) {
  constexpr int FPL = K / 32;  // 2 for K=64, 4 for K=128
  int tid  = blockIdx.x * blockDim.x + threadIdx.x;
  int e    = tid >> 5;
  int lane = tid & 31;
  if (e >= EE) return;
  int s = src[e];
  int d = dst[e];
  const float* xs = x + (size_t)s * K + lane * FPL;
  float*       ad = agg + (size_t)d * K + lane * FPL;
  if constexpr (FPL == 4) {
    float4 v = *(const float4*)xs;
    unsafeAtomicAdd(&ad[0], v.x);
    unsafeAtomicAdd(&ad[1], v.y);
    unsafeAtomicAdd(&ad[2], v.z);
    unsafeAtomicAdd(&ad[3], v.w);
  } else {
    float2 v = *(const float2*)xs;
    unsafeAtomicAdd(&ad[0], v.x);
    unsafeAtomicAdd(&ad[1], v.y);
  }
}

// ---------------- h = (1+eps)*x + agg -> bf16 staging (8 elems/thread) ----------------
template <int K>
__global__ void combine_bf16(const float* __restrict__ cur, const float* __restrict__ agg,
                             const float* __restrict__ epsP, ushort_t* __restrict__ abf) {
  float ep  = 1.0f + *epsP;
  size_t n8 = (size_t)NN * K / 8;
  for (size_t i = (size_t)blockIdx.x * blockDim.x + threadIdx.x; i < n8;
       i += (size_t)gridDim.x * blockDim.x) {
    const float4* c4 = (const float4*)cur + i * 2;
    const float4* a4 = (const float4*)agg + i * 2;
    float4 x0 = c4[0], x1 = c4[1];
    float4 g0 = a4[0], g1 = a4[1];
    uint4 o;
    o.x = pack2bf(fmaf(ep, x0.x, g0.x), fmaf(ep, x0.y, g0.y));
    o.y = pack2bf(fmaf(ep, x0.z, g0.z), fmaf(ep, x0.w, g0.w));
    o.z = pack2bf(fmaf(ep, x1.x, g1.x), fmaf(ep, x1.y, g1.y));
    o.w = pack2bf(fmaf(ep, x1.z, g1.z), fmaf(ep, x1.w, g1.w));
    ((uint4*)abf)[i] = o;
  }
}

// ---------------- vectorized f32 -> bf16 conversion (n multiple of 8) ----------------
__global__ void conv_bf16_vec(const float* __restrict__ w, ushort_t* __restrict__ o, size_t n8) {
  for (size_t i = (size_t)blockIdx.x * blockDim.x + threadIdx.x; i < n8;
       i += (size_t)gridDim.x * blockDim.x) {
    const float4* w4 = (const float4*)w + i * 2;
    float4 x0 = w4[0], x1 = w4[1];
    uint4 ov;
    ov.x = pack2bf(x0.x, x0.y);
    ov.y = pack2bf(x0.z, x0.w);
    ov.z = pack2bf(x1.x, x1.y);
    ov.w = pack2bf(x1.z, x1.w);
    ((uint4*)o)[i] = ov;
  }
}

// ---------------- pack weights into WMMA B-operand lane layout ----------------
// fragment u = ks*8 + tile lives at bp + u*32*16; element e -> B[ks*32+hi*8+e+(e>=8?8:0)][tile*16+l]
template <int K>
__global__ void pack_b(const float* __restrict__ w, ushort_t* __restrict__ bp) {
  constexpr int KSTEPS = K / 32;
  int idx = blockIdx.x * blockDim.x + threadIdx.x;
  if (idx >= 8 * KSTEPS * 32) return;
  int lane = idx & 31;
  int rest = idx >> 5;  // rest = ks*8 + t
  int t    = rest & 7;
  int ks   = rest >> 3;
  int l    = lane & 15;
  int hi   = lane >> 4;
  int col  = t * 16 + l;
  int k0   = ks * 32 + hi * 8;
  ushort_t* o = bp + (size_t)idx * 16;
#pragma unroll
  for (int e = 0; e < 16; ++e) o[e] = f2bf(w[(size_t)(k0 + e + (e >= 8 ? 8 : 0)) * HH + col]);
}

// ---------------- WMMA GEMM: out[N x 128] = A[N x K](bf16) @ B[K x 128](packed bf16) + bias --------
// block = 256 threads = 8 waves covering a 128-row stripe; wave w -> rows [w*16, w*16+16),
// iterates all 8 column tiles holding 8 v8f accumulators; A/B fragments double-buffered so each
// WMMA's operands are loaded one iteration ahead (load latency overlaps prior WMMA).
template <int K, bool RELU>
__global__ void __launch_bounds__(256) gemm_bf16(const ushort_t* __restrict__ A,
                                                 const ushort_t* __restrict__ Bp,
                                                 const float* __restrict__ bias,
                                                 float* __restrict__ out) {
  constexpr int KSTEPS = K / 32;
  constexpr int TOT    = KSTEPS * 8;
  int wave = threadIdx.x >> 5;
  int lane = threadIdx.x & 31;
  int l    = lane & 15;
  int hi   = lane >> 4;
  int rowBase = blockIdx.x * 128 + wave * 16;
  if (rowBase >= NN) return;

  v8f c[8];
#pragma unroll
  for (int t = 0; t < 8; ++t) {
    float bv = bias[t * 16 + l];
#pragma unroll
    for (int r = 0; r < 8; ++r) c[t][r] = bv;
  }

  const ushort_t* arow = A + (size_t)(rowBase + l) * K;
  const ushort_t* bbase = Bp + (size_t)lane * 16;

  BfVec a[2], b[2];
  // A 16x32 bf16 fragment (ISA 7.12.2): lane<16 -> K[kb..+7],K[kb+16..+23]; lane>=16 -> +8
  a[0].q[0] = *(const uint4*)(arow + hi * 8);
  a[0].q[1] = *(const uint4*)(arow + 16 + hi * 8);
  b[0].q[0] = ((const uint4*)bbase)[0];
  b[0].q[1] = ((const uint4*)bbase)[1];

#pragma unroll
  for (int u = 0; u < TOT; ++u) {
    int ks = u >> 3;
    int t  = u & 7;
    if (t == 0 && ks + 1 < KSTEPS) {
      int kb = (ks + 1) * 32;
      a[(ks + 1) & 1].q[0] = *(const uint4*)(arow + kb + hi * 8);
      a[(ks + 1) & 1].q[1] = *(const uint4*)(arow + kb + 16 + hi * 8);
    }
    if (u + 1 < TOT) {
      const uint4* bp = (const uint4*)(bbase + (size_t)(u + 1) * 32 * 16);
      b[(u + 1) & 1].q[0] = bp[0];
      b[(u + 1) & 1].q[1] = bp[1];
    }
    c[t] = __builtin_amdgcn_wmma_f32_16x16x32_bf16(false, a[ks & 1].v, false, b[u & 1].v,
                                                   (short)0, c[t], false, false);
  }

  // C/D layout: VGPR r, lane<16 -> (M=r, N=l); lane>=16 -> (M=r+8, N=l)
#pragma unroll
  for (int t = 0; t < 8; ++t) {
    float* orow = out + (size_t)(rowBase + hi * 8) * HH + t * 16 + l;
#pragma unroll
    for (int r = 0; r < 8; ++r) {
      float v = c[t][r];
      if (RELU) v = fmaxf(v, 0.0f);
      orow[(size_t)r * HH] = v;
    }
  }
}

// ---------------- batch-norm statistics: per-feature sum / sumsq ----------------
__global__ void __launch_bounds__(128) bn_stats(const float* __restrict__ t,
                                                float* __restrict__ stats) {
  int f  = threadIdx.x;
  int r0 = blockIdx.x * 256;
  int r1 = r0 + 256;
  if (r1 > NN) r1 = NN;
  float s = 0.0f, s2 = 0.0f;
  for (int r = r0; r < r1; ++r) {
    float v = t[(size_t)r * HH + f];
    s += v;
    s2 = fmaf(v, v, s2);
  }
  unsafeAtomicAdd(&stats[f], s);
  unsafeAtomicAdd(&stats[HH + f], s2);
}

__global__ void bn_final(float* __restrict__ stats, const float* __restrict__ g,
                         const float* __restrict__ be) {
  int f = threadIdx.x;
  float mu  = stats[f] * (1.0f / NN);
  float var = stats[HH + f] * (1.0f / NN) - mu * mu;
  float sc  = g[f] * rsqrtf(var + 1e-5f);
  stats[2 * HH + f] = sc;
  stats[3 * HH + f] = be[f] - mu * sc;
}

// ---------------- BN apply + relu -> bf16 staging (8 elems/thread) ----------------
__global__ void bn_relu_bf16(const float* __restrict__ t, const float* __restrict__ stats,
                             ushort_t* __restrict__ abf) {
  size_t n8 = (size_t)NN * HH / 8;
  for (size_t i = (size_t)blockIdx.x * blockDim.x + threadIdx.x; i < n8;
       i += (size_t)gridDim.x * blockDim.x) {
    int fb = (int)((i * 8) & (HH - 1));
    const float4* sc4 = (const float4*)(stats + 2 * HH + fb);
    const float4* sh4 = (const float4*)(stats + 3 * HH + fb);
    float4 s0 = sc4[0], s1 = sc4[1];
    float4 h0 = sh4[0], h1 = sh4[1];
    const float4* t4 = (const float4*)t + i * 2;
    float4 x0 = t4[0], x1 = t4[1];
    float v0 = fmaxf(fmaf(x0.x, s0.x, h0.x), 0.0f);
    float v1 = fmaxf(fmaf(x0.y, s0.y, h0.y), 0.0f);
    float v2 = fmaxf(fmaf(x0.z, s0.z, h0.z), 0.0f);
    float v3 = fmaxf(fmaf(x0.w, s0.w, h0.w), 0.0f);
    float v4 = fmaxf(fmaf(x1.x, s1.x, h1.x), 0.0f);
    float v5 = fmaxf(fmaf(x1.y, s1.y, h1.y), 0.0f);
    float v6 = fmaxf(fmaf(x1.z, s1.z, h1.z), 0.0f);
    float v7 = fmaxf(fmaf(x1.w, s1.w, h1.w), 0.0f);
    uint4 o;
    o.x = pack2bf(v0, v1);
    o.y = pack2bf(v2, v3);
    o.z = pack2bf(v4, v5);
    o.w = pack2bf(v6, v7);
    ((uint4*)abf)[i] = o;
  }
}

// ---------------- head: out[N x 2] = t[N x 128] @ hwb[128 x 2] + hbb ----------------
__global__ void head_out(const float* __restrict__ t, const float* __restrict__ hwb,
                         const float* __restrict__ hbb, float* __restrict__ out) {
  int r = blockIdx.x * blockDim.x + threadIdx.x;
  if (r >= NN) return;
  float a0 = hbb[0], a1 = hbb[1];
  const float4* row4 = (const float4*)(t + (size_t)r * HH);
  const float4* w4   = (const float4*)hwb;  // 128x2 row-major: float4 = 2 k-rows
#pragma unroll 8
  for (int k4 = 0; k4 < HH / 4; ++k4) {
    float4 v  = row4[k4];
    float4 w0 = w4[2 * k4];      // rows 4k, 4k+1
    float4 w1 = w4[2 * k4 + 1];  // rows 4k+2, 4k+3
    a0 = fmaf(v.x, w0.x, fmaf(v.y, w0.z, fmaf(v.z, w1.x, fmaf(v.w, w1.z, a0))));
    a1 = fmaf(v.x, w0.y, fmaf(v.y, w0.w, fmaf(v.z, w1.y, fmaf(v.w, w1.w, a1))));
  }
  out[2 * r]     = a0;
  out[2 * r + 1] = a1;
}

extern "C" void kernel_launch(void* const* d_in, const int* in_sizes, int n_in,
                              void* d_out, int out_size, void* d_ws, size_t ws_size,
                              hipStream_t stream) {
  (void)in_sizes; (void)n_in; (void)out_size; (void)ws_size;

  const float* x   = (const float*)d_in[0];
  const int*   ei  = (const int*)d_in[1];
  const int*   src = ei;
  const int*   dst = ei + EE;
  const float *wa[3], *ba[3], *g[3], *be[3], *wb[3], *bb[3], *eps[3];
  for (int l = 0; l < 3; ++l) {
    int o  = 2 + 7 * l;
    wa[l]  = (const float*)d_in[o + 0];
    ba[l]  = (const float*)d_in[o + 1];
    g[l]   = (const float*)d_in[o + 2];
    be[l]  = (const float*)d_in[o + 3];
    wb[l]  = (const float*)d_in[o + 4];
    bb[l]  = (const float*)d_in[o + 5];
    eps[l] = (const float*)d_in[o + 6];
  }
  const float* hwa = (const float*)d_in[23];
  const float* hba = (const float*)d_in[24];
  const float* hwb = (const float*)d_in[25];
  const float* hbb = (const float*)d_in[26];

  // workspace layout: B0, B1 (N x 128 f32), ABF (N x 128 bf16), packed weights, BN stats
  float*    B0    = (float*)d_ws;
  float*    B1    = B0 + (size_t)NN * HH;
  ushort_t* ABF   = (ushort_t*)(B1 + (size_t)NN * HH);
  ushort_t* WAB   = ABF + (size_t)NN * HH;   // packed B for first lin (<= 128x128)
  ushort_t* WBB   = WAB + HH * HH;           // packed B for second lin
  float*    STATS = (float*)(WBB + HH * HH);

  const int TPB      = 256;
  const int gemmGrid = (NN + 127) / 128;  // 782 stripes of 128 rows
  const int elGrid   = 2048;
  const int statGrid = (NN + 255) / 256;

  const float* cur = x;
  for (int l = 0; l < 3; ++l) {
    int    K    = (l == 0) ? DIN_ : HH;
    float* work = (l == 2) ? B1 : B0;    // agg, then GEMM1 tmp
    float* outb = (l == 0) ? B1 : work;  // layer output (work is dead by then)

    hipMemsetAsync(work, 0, (size_t)NN * K * sizeof(float), stream);
    hipMemsetAsync(STATS, 0, 2 * HH * sizeof(float), stream);

    if (K == DIN_) {
      pack_b<DIN_><<<1, TPB, 0, stream>>>(wa[l], WAB);
      pack_b<HH><<<4, TPB, 0, stream>>>(wb[l], WBB);
      scatter_agg<DIN_><<<EE / 8, TPB, 0, stream>>>(cur, src, dst, work);
      combine_bf16<DIN_><<<elGrid, TPB, 0, stream>>>(cur, work, eps[l], ABF);
      gemm_bf16<DIN_, false><<<gemmGrid, TPB, 0, stream>>>(ABF, WAB, ba[l], work);
    } else {
      pack_b<HH><<<4, TPB, 0, stream>>>(wa[l], WAB);
      pack_b<HH><<<4, TPB, 0, stream>>>(wb[l], WBB);
      scatter_agg<HH><<<EE / 8, TPB, 0, stream>>>(cur, src, dst, work);
      combine_bf16<HH><<<elGrid, TPB, 0, stream>>>(cur, work, eps[l], ABF);
      gemm_bf16<HH, false><<<gemmGrid, TPB, 0, stream>>>(ABF, WAB, ba[l], work);
    }

    bn_stats<<<statGrid, HH, 0, stream>>>(work, STATS);
    bn_final<<<1, HH, 0, stream>>>(STATS, g[l], be[l]);
    bn_relu_bf16<<<elGrid, TPB, 0, stream>>>(work, STATS, ABF);
    gemm_bf16<HH, true><<<gemmGrid, TPB, 0, stream>>>(ABF, WBB, bb[l], outb);
    cur = outb;  // L0: B1, L1: B0, L2: B1
  }

  // head MLP: relu(cur @ hwa + hba) @ hwb + hbb
  pack_b<HH><<<4, TPB, 0, stream>>>(hwa, WAB);
  conv_bf16_vec<<<elGrid, TPB, 0, stream>>>(cur, ABF, (size_t)NN * HH / 8);
  gemm_bf16<HH, true><<<gemmGrid, TPB, 0, stream>>>(ABF, WAB, hba, B0);
  head_out<<<(NN + 255) / 256, TPB, 0, stream>>>(B0, hwb, hbb, (float*)d_out);
}
